// StackedGCNEncoder_89343909692103
// MI455X (gfx1250) — matrix operand order repaired and checked by reading, not compiled, not run on hardware
//
#include <hip/hip_runtime.h>

typedef __attribute__((ext_vector_type(16))) _Float16 v16h;
typedef __attribute__((ext_vector_type(8)))  float    v8f;

// ---------------------------------------------------------------- utilities

__global__ void fill_f32(float* __restrict__ p, float v, long long n) {
  long long i = (long long)blockIdx.x * blockDim.x + threadIdx.x;
  long long stride = (long long)gridDim.x * blockDim.x;
  for (; i < n; i += stride) p[i] = v;
}

// deg starts at 1.0 (self loop); count incoming edges per destination.
__global__ void deg_count(const int* __restrict__ col, float* __restrict__ deg,
                          int E, int N) {
  int i = blockIdx.x * blockDim.x + threadIdx.x;
  int stride = gridDim.x * blockDim.x;
  for (; i < E; i += stride) {
    int c = col[i];
    if ((unsigned)c < (unsigned)N) atomicAdd(&deg[c], 1.0f);
  }
}

__global__ void deg_to_dinv(float* __restrict__ deg, int N) {
  int i = blockIdx.x * blockDim.x + threadIdx.x;
  if (i < N) {
    float d = deg[i];
    deg[i] = (d > 0.0f) ? rsqrtf(d) : 0.0f;
  }
}

// ---------------------------------------------------------------- WMMA GEMM
// Y[M x NOUT] = X[M x K] * W[K x NOUT], f32 in/out, f16 multiply, f32 accum.
// Each wave owns one 16-wide N tile: it loads its K x 16 panel of W into
// registers ONCE (v16h per 32-K step), then sweeps MT row-tiles of 16,
// issuing K/32 v_wmma_f32_16x16x32_f16 per tile straight out of registers.
// blockDim.x = 32 * (NOUT/16); grid.x = ceil(M / (16*MT)).
template <int K, int NOUT, int MT>
__global__ void gemm_wmma_f16(const float* __restrict__ X,
                              const float* __restrict__ W,
                              float* __restrict__ Y, int M) {
  const int lane  = threadIdx.x & 31;
  const int wave  = threadIdx.x >> 5;   // which 16-wide N tile
  const int lrow  = lane & 15;
  const int khalf = lane >> 4;          // 0 or 1
  const int ncol  = wave * 16 + lrow;

  // B 32x16 f16 per k-step: lanes 0-15 hold K = 32s+{0..7,16..23} of column N=lane;
  //                         lanes 16-31 hold K = 32s+{8..15,24..31} of column N=lane-16.
  v16h b[K / 32];
#pragma unroll
  for (int s = 0; s < K / 32; ++s)
#pragma unroll
    for (int j = 0; j < 16; ++j)
      b[s][j] = (_Float16)W[(size_t)(32 * s + 8 * khalf + (j & 7) + ((j >> 3) << 4)) * NOUT + ncol];

  const int mbase = blockIdx.x * (16 * MT);
#pragma unroll
  for (int t = 0; t < MT; ++t) {
    const int m0 = mbase + 16 * t;
    if (m0 >= M) break;                       // uniform; EXEC untouched
    int mrow = m0 + lrow;
    if (mrow >= M) mrow = M - 1;              // clamp, keep EXEC all-ones
    const float* xr = X + (size_t)mrow * K;

    v8f acc = {};
#pragma unroll
    for (int s = 0; s < K / 32; ++s) {
      // A 16x32 f16: lanes 0-15 hold M=lane, K = 32s+{0..7,16..23};
      //              lanes 16-31 hold M=lane-16, K = 32s+{8..15,24..31}.
      v16h a;
#pragma unroll
      for (int j = 0; j < 8; ++j) {
        a[j]     = (_Float16)xr[32 * s + 8 * khalf + j];
        a[j + 8] = (_Float16)xr[32 * s + 16 + 8 * khalf + j];
      }
      acc = __builtin_amdgcn_wmma_f32_16x16x32_f16(
          /*neg_a=*/false, a, /*neg_b=*/false, b[s],
          /*c_mod=*/(short)0, acc, /*reuse_a=*/false, /*reuse_b=*/false);
    }

    // C/D: VGPR r holds M = m0 + 8*khalf + r, N = ncol.
    if (m0 + 16 <= M) {                       // fast path: no per-row guards
#pragma unroll
      for (int r = 0; r < 8; ++r)
        Y[(size_t)(m0 + 8 * khalf + r) * NOUT + ncol] = acc[r];
    } else {
#pragma unroll
      for (int r = 0; r < 8; ++r) {
        int m = m0 + 8 * khalf + r;
        if (m < M) Y[(size_t)m * NOUT + ncol] = acc[r];
      }
    }
  }
}

// ---------------------------------------------------------------- scatter
// dst[col[e]] += src[row[e]] * dinv[row[e]]*dinv[col[e]]
// One thread per (edge, float4 chunk): consecutive lanes share one edge and
// touch consecutive addresses; atomic conflicts rare on a random graph.
template <int F>
__global__ void scatter_add(const int* __restrict__ row,
                            const int* __restrict__ col,
                            const float* __restrict__ dinv,
                            const float* __restrict__ src,
                            float* __restrict__ dst, int E, int N) {
  const int CH = F / 4;
  long long idx = (long long)blockIdx.x * blockDim.x + threadIdx.x;
  long long stride = (long long)gridDim.x * blockDim.x;
  const long long total = (long long)E * CH;
  for (; idx < total; idx += stride) {
    int e  = (int)(idx / CH);
    int fc = (int)(idx - (long long)e * CH) * 4;
    int r = row[e], c = col[e];
    if ((unsigned)r >= (unsigned)N || (unsigned)c >= (unsigned)N) continue;
    float nrm = dinv[r] * dinv[c];
    const float4 v = *reinterpret_cast<const float4*>(src + (size_t)r * F + fc);
    float* d = dst + (size_t)c * F + fc;
    atomicAdd(d + 0, v.x * nrm);
    atomicAdd(d + 1, v.y * nrm);
    atomicAdd(d + 2, v.z * nrm);
    atomicAdd(d + 3, v.w * nrm);
  }
}

// ---------------------------------------------------------------- epilogue
// acc[n,f] = (acc[n,f] + dinv[n]^2 * selfmsg[n,f] + bias[f])  [, relu]
template <int F, bool RELU>
__global__ void finalize(float* __restrict__ acc,
                         const float* __restrict__ selfmsg,
                         const float* __restrict__ dinv,
                         const float* __restrict__ bias, int N) {
  long long idx = (long long)blockIdx.x * blockDim.x + threadIdx.x;
  long long stride = (long long)gridDim.x * blockDim.x;
  const long long total = (long long)N * F;
  for (; idx < total; idx += stride) {
    int n = (int)(idx / F);
    int f = (int)(idx - (long long)n * F);
    float di = dinv[n];
    float v = acc[idx] + di * di * selfmsg[idx] + bias[f];
    if (RELU) v = fmaxf(v, 0.0f);
    acc[idx] = v;
  }
}

// ---------------------------------------------------------------- launcher

extern "C" void kernel_launch(void* const* d_in, const int* in_sizes, int n_in,
                              void* d_out, int out_size, void* d_ws, size_t ws_size,
                              hipStream_t stream) {
  const float* x  = (const float*)d_in[0];
  const int*   ei = (const int*)d_in[1];   // edge_index [2, E]
  const float* W1 = (const float*)d_in[2];
  const float* b1 = (const float*)d_in[3];
  const float* W2 = (const float*)d_in[4];
  const float* b2 = (const float*)d_in[5];

  const int DH   = in_sizes[3];                 // 64
  const int DIN  = in_sizes[2] / DH;            // 64
  const int DOUT = in_sizes[5];                 // 32
  const int N    = in_sizes[0] / DIN;           // 100000
  const int E    = in_sizes[1] / 2;             // 1600000
  (void)n_in; (void)out_size; (void)ws_size; (void)DOUT;

  const int* row = ei;
  const int* col = ei + E;

  // workspace layout (floats): dinv | hW1 | agg1(h1) | hW2
  float* dinv = (float*)d_ws;                       // N
  float* hW1  = dinv + N;                           // N*64
  float* agg1 = hW1 + (size_t)N * 64;               // N*64
  float* hW2  = agg1 + (size_t)N * 64;              // N*32
  float* out  = (float*)d_out;                      // N*32

  constexpr int MT = 8;                             // 16*8 = 128 rows per block
  const int gblocks = (N + 16 * MT - 1) / (16 * MT);

  // --- degrees (self loop included) -> dinv
  fill_f32<<<512, 256, 0, stream>>>(dinv, 1.0f, (long long)N);
  deg_count<<<(E + 255) / 256, 256, 0, stream>>>(col, dinv, E, N);
  deg_to_dinv<<<(N + 255) / 256, 256, 0, stream>>>(dinv, N);

  // --- layer 1: hW1 = x @ W1 (WMMA), scatter, self-loop + bias + relu
  gemm_wmma_f16<64, 64, MT><<<gblocks, 4 * 32, 0, stream>>>(x, W1, hW1, N);
  fill_f32<<<2048, 256, 0, stream>>>(agg1, 0.0f, (long long)N * 64);
  {
    long long total = (long long)E * (64 / 4);
    int blocks = (int)((total + 255) / 256);
    scatter_add<64><<<blocks, 256, 0, stream>>>(row, col, dinv, hW1, agg1, E, N);
  }
  finalize<64, true><<<2048, 256, 0, stream>>>(agg1, hW1, dinv, b1, N);

  // --- layer 2: hW2 = h1 @ W2 (WMMA), scatter into d_out, self-loop + bias
  gemm_wmma_f16<64, 32, MT><<<gblocks, 2 * 32, 0, stream>>>(agg1, W2, hW2, N);
  fill_f32<<<2048, 256, 0, stream>>>(out, 0.0f, (long long)N * 32);
  {
    long long total = (long long)E * (32 / 4);
    int blocks = (int)((total + 255) / 256);
    scatter_add<32><<<blocks, 256, 0, stream>>>(row, col, dinv, hW2, out, E, N);
  }
  finalize<32, false><<<2048, 256, 0, stream>>>(out, hW2, dinv, b2, N);
}
